// Dynamic_37933151158962
// MI455X (gfx1250) — matrix operand order repaired and checked by reading, not compiled, not run on hardware
//
#include <hip/hip_runtime.h>
#include <stdint.h>

#define Bdim  64
#define Tdim  64
#define Adim  8
#define Gdim  128
#define GPdim 256
#define SVdim 512
#define SPdim 256
#define SHdim 512
#define G3dim 384
#define GGdim 16384            // G*G
#define BT    (Bdim*Tdim)      // 4096
#define OUTC  (2*Gdim + 2*SPdim) // 768

typedef __attribute__((ext_vector_type(16))) __bf16 v16bf;
typedef __attribute__((ext_vector_type(8)))  __bf16 v8bf;
typedef __attribute__((ext_vector_type(8)))  float  v8f;

__device__ __forceinline__ __bf16 f2bf(float f) {
    union { float f; uint32_t u; } a; a.f = f;
    uint32_t r = a.u + 0x7FFFu + ((a.u >> 16) & 1u);   // round-to-nearest-even
    uint16_t h = (uint16_t)(r >> 16);
    return __builtin_bit_cast(__bf16, h);
}
__device__ __forceinline__ float bf2f(__bf16 b) {
    uint16_t h = __builtin_bit_cast(uint16_t, b);
    union { uint32_t u; float f; } a; a.u = ((uint32_t)h) << 16;
    return a.f;
}
__device__ __forceinline__ float eluf(float x) { return x > 0.f ? x : (expf(x) - 1.f); }
__device__ __forceinline__ float sigm(float x) { return 1.f / (1.f + expf(-x)); }

// ---------------------------------------------------------------------------
// f32 -> bf16 elementwise convert
__global__ __launch_bounds__(256) void k_cvt_bf16(const float* __restrict__ in,
                                                  __bf16* __restrict__ out, int n) {
    int i = blockIdx.x * 256 + threadIdx.x;
    if (i < n) out[i] = f2bf(in[i]);
}

// Transpose-convert: W (K x N, f32) -> Wt (N x K, bf16). idx is coalesced on write.
__global__ __launch_bounds__(256) void k_transpose_bf16(const float* __restrict__ W,
                                                        __bf16* __restrict__ Wt,
                                                        int K, int N) {
    int idx = blockIdx.x * 256 + threadIdx.x;
    if (idx >= K * N) return;
    int n = idx / K, k = idx - n * K;
    Wt[idx] = f2bf(W[(size_t)k * N + n]);
}

// ---------------------------------------------------------------------------
// bf16 WMMA GEMM: C[M,N] = A[M,K] @ B[K,N] (+bias), with B supplied transposed
// as Bt[N,K].  Block = 4 waves, each wave computes one 16x64 strip (4 WMMA
// accumulators reusing the A fragment); block tile = 64x64.
// Output: f32 to Cf, or bf16 to Cb (whichever is non-null).
__global__ __launch_bounds__(128) void k_gemm_bf16(
    const __bf16* __restrict__ A, const __bf16* __restrict__ Bt,
    const float* __restrict__ bias,
    float* __restrict__ Cf, __bf16* __restrict__ Cb,
    int M, int N, int K)
{
    const int lane = threadIdx.x & 31;
    const int wave = threadIdx.x >> 5;
    const int half = lane >> 4;      // K-half selector (contiguous k = 16*half + e)
    const int lm   = lane & 15;      // row (A) / col (B,C) within tile
    const int row0 = blockIdx.x * 64 + wave * 16;
    const int col0 = blockIdx.y * 64;

    v8f acc0 = {}, acc1 = {}, acc2 = {}, acc3 = {};

    const __bf16* ap  = A  + (size_t)(row0 + lm) * K + half * 16;
    const __bf16* bp0 = Bt + (size_t)(col0 +  0 + lm) * K + half * 16;
    const __bf16* bp1 = Bt + (size_t)(col0 + 16 + lm) * K + half * 16;
    const __bf16* bp2 = Bt + (size_t)(col0 + 32 + lm) * K + half * 16;
    const __bf16* bp3 = Bt + (size_t)(col0 + 48 + lm) * K + half * 16;

    for (int kb = 0; kb < K; kb += 32) {
        union { v16bf v; v8bf h[2]; } a, b0, b1, b2, b3;
        // A fragment: elem e -> K = kb + half*16 + e (contiguous in A row)
        a.h[0] = *(const v8bf*)(ap + kb);
        a.h[1] = *(const v8bf*)(ap + kb + 8);
        // B fragment: elem e -> K = kb + half*16 + e (contiguous in Bt row)
        b0.h[0] = *(const v8bf*)(bp0 + kb); b0.h[1] = *(const v8bf*)(bp0 + kb + 8);
        b1.h[0] = *(const v8bf*)(bp1 + kb); b1.h[1] = *(const v8bf*)(bp1 + kb + 8);
        b2.h[0] = *(const v8bf*)(bp2 + kb); b2.h[1] = *(const v8bf*)(bp2 + kb + 8);
        b3.h[0] = *(const v8bf*)(bp3 + kb); b3.h[1] = *(const v8bf*)(bp3 + kb + 8);
        if (kb + 32 < K) {                       // global_prefetch_b8 next chunk
            __builtin_prefetch(ap  + kb + 32, 0, 1);
            __builtin_prefetch(bp0 + kb + 32, 0, 1);
            __builtin_prefetch(bp2 + kb + 32, 0, 1);
        }
        acc0 = __builtin_amdgcn_wmma_f32_16x16x32_bf16(false, a.v, false, b0.v, (short)0, acc0, false, false);
        acc1 = __builtin_amdgcn_wmma_f32_16x16x32_bf16(false, a.v, false, b1.v, (short)0, acc1, false, false);
        acc2 = __builtin_amdgcn_wmma_f32_16x16x32_bf16(false, a.v, false, b2.v, (short)0, acc2, false, false);
        acc3 = __builtin_amdgcn_wmma_f32_16x16x32_bf16(false, a.v, false, b3.v, (short)0, acc3, false, false);
    }

    // C/D layout: VGPR r, lane L -> row = row0 + 8*half + r, col = col0 + 16*nt + lm
    v8f accs[4] = { acc0, acc1, acc2, acc3 };
    #pragma unroll
    for (int nt = 0; nt < 4; ++nt) {
        const int col = col0 + nt * 16 + lm;
        const float bv = bias ? bias[col] : 0.f;
        #pragma unroll
        for (int r = 0; r < 8; ++r) {
            const int row = row0 + half * 8 + r;
            const float val = accs[nt][r] + bv;
            if (Cb) Cb[(size_t)row * N + col] = f2bf(val);
            else    Cf[(size_t)row * N + col] = val;
        }
    }
}

// ---------------------------------------------------------------------------
// LayerNorm + ELU over rows of h (BT x 256), bf16 output. One block per row.
__global__ __launch_bounds__(256) void k_ln_elu(const float* __restrict__ h,
                                                const float* __restrict__ g,
                                                const float* __restrict__ bb,
                                                __bf16* __restrict__ out) {
    __shared__ float sred[256];
    const int row = blockIdx.x, tid = threadIdx.x;
    const float x = h[(size_t)row * SPdim + tid];
    sred[tid] = x; __syncthreads();
    for (int s = 128; s > 0; s >>= 1) { if (tid < s) sred[tid] += sred[tid + s]; __syncthreads(); }
    const float mean = sred[0] * (1.f / SPdim);
    __syncthreads();
    const float d = x - mean;
    sred[tid] = d * d; __syncthreads();
    for (int s = 128; s > 0; s >>= 1) { if (tid < s) sred[tid] += sred[tid + s]; __syncthreads(); }
    const float var = sred[0] * (1.f / SPdim);
    const float y = d * rsqrtf(var + 1e-5f) * g[tid] + bb[tid];
    out[(size_t)row * SPdim + tid] = f2bf(eluf(y));
}

// ---------------------------------------------------------------------------
// h2 = tanh(act @ Wa1) : (BT x 8) @ (8 x 384) -> bf16 (BT x 384)
__global__ __launch_bounds__(128) void k_tanh_a(const float* __restrict__ act,
                                                const float* __restrict__ Wa1,
                                                __bf16* __restrict__ h2) {
    const int j = blockIdx.y * 128 + threadIdx.x;
    const int r = blockIdx.x;
    if (j >= G3dim) return;
    float s = 0.f;
    #pragma unroll
    for (int k = 0; k < Adim; ++k) s = fmaf(act[(size_t)r * Adim + k], Wa1[(size_t)k * G3dim + j], s);
    h2[(size_t)r * G3dim + j] = f2bf(tanhf(s));
}

// ---------------------------------------------------------------------------
// Block-wide helpers for the scan kernel (blockDim == 256, caller syncs).
__device__ void vecmat(const float* __restrict__ v, const float* __restrict__ W,
                       const float* __restrict__ bias, float* __restrict__ o,
                       int In, int Out) {
    for (int j = threadIdx.x; j < Out; j += 256) {
        float s = bias ? bias[j] : 0.f;
        #pragma unroll 4
        for (int i = 0; i < In; ++i) s = fmaf(v[i], W[(size_t)i * Out + j], s);
        o[j] = s;
    }
}
__device__ void retrv(const float* __restrict__ q, const float* __restrict__ M,
                      float* __restrict__ o) {
    for (int e = threadIdx.x; e < SPdim; e += 256) {
        float s = 0.f;
        #pragma unroll 4
        for (int d = 0; d < SPdim; ++d) s = fmaf(q[d], M[(size_t)d * SPdim + e], s);
        o[e] = s;
    }
}

// Persistent scan: one block per batch, T=64 steps, M/S state L2-resident.
__global__ __launch_bounds__(256) void k_scan(
    const __bf16* __restrict__ Wt, const float* __restrict__ xs,
    const float* __restrict__ g0,  const float* __restrict__ Wp,
    const float* __restrict__ Wmu1, const float* __restrict__ bmu1,
    const float* __restrict__ Wmu2, const float* __restrict__ bmu2,
    const float* __restrict__ Wd1,  const float* __restrict__ bd1,
    const float* __restrict__ Wd2,  const float* __restrict__ bd2,
    const float* __restrict__ Wx1,  const float* __restrict__ bx1,
    const float* __restrict__ Wx2,  const float* __restrict__ bx2,
    const float* __restrict__ Wk,   const float* __restrict__ Wv,
    const float* __restrict__ M0,   const float* __restrict__ S0,
    float* __restrict__ Mw, float* __restrict__ Sw, float* __restrict__ out)
{
    __shared__ float sg[Gdim], sggen[Gdim], sgint[Gdim], sgmu[Gdim], sgate[Gdim];
    __shared__ float sgi[2 * Gdim + 1];
    __shared__ float sgproj[SPdim], sxp[SPdim], sq[SPdim], sr[SPdim], sxrp[SPdim];
    __shared__ float sk[SPdim], sv2[SPdim], srk[SPdim], smin[SPdim];
    __shared__ float sh[10 * Gdim];   // 1280: max hidden (gate mlp)
    __shared__ float sred[256];

    const int b = blockIdx.x, tid = threadIdx.x;
    float* M = Mw + (size_t)b * SPdim * SPdim;
    float* S = Sw + (size_t)b * SPdim * SPdim;

    for (int i = tid; i < SPdim * SPdim; i += 256) {        // don't mutate inputs
        M[i] = M0[(size_t)b * SPdim * SPdim + i];
        S[i] = S0[(size_t)b * SPdim * SPdim + i];
    }
    if (tid < Gdim) sg[tid] = g0[tid];
    __syncthreads();

    for (int t = 0; t < Tdim; ++t) {
        sxp[tid] = xs[((size_t)b * Tdim + t) * SPdim + tid];
        __syncthreads();
        // grid_step with hoisted Wt
        if (tid < Gdim) {
            const __bf16* wr = Wt + ((size_t)b * Tdim + t) * GGdim + (size_t)tid * Gdim;
            float s = 0.f;
            for (int j = 0; j < Gdim; ++j) s = fmaf(bf2f(wr[j]), sg[j], s);
            sggen[tid] = fminf(10.f, fmaxf(-10.f, sg[tid] + s));
        }
        __syncthreads();

        float* orow = out + ((size_t)b * Tdim + t) * OUTC;

        if (t == 0) {
            if (tid < Gdim) { orow[tid] = sggen[tid]; orow[Gdim + tid] = sggen[tid]; }
            orow[2 * Gdim + tid]         = sxp[tid];
            orow[2 * Gdim + SPdim + tid] = sxp[tid];
            vecmat(sggen, Wp, nullptr, sgproj, Gdim, SPdim);
            __syncthreads();
            smin[tid] = sgproj[tid] + sxp[tid];
            if (tid < Gdim) sg[tid] = sggen[tid];
            __syncthreads();
        } else {
            vecmat(sggen, Wp, nullptr, sgproj, Gdim, SPdim);
            __syncthreads();
            // x_gen_pred = mlp_x(retrieve(g_proj, M)) -> out[xg]
            retrv(sgproj, M, sr); __syncthreads();
            vecmat(sr, Wx1, bx1, sh, SPdim, SHdim); __syncthreads();
            for (int j = tid; j < SHdim; j += 256) sh[j] = eluf(sh[j]);
            __syncthreads();
            vecmat(sh, Wx2, bx2, orow + 2 * Gdim, SHdim, SPdim); __syncthreads();
            // gx_retr = retrieve(g_proj + xp, M)
            sq[tid] = sgproj[tid] + sxp[tid]; __syncthreads();
            retrv(sq, M, sr); __syncthreads();
            // x_retr_pred
            vecmat(sr, Wx1, bx1, sh, SPdim, SHdim); __syncthreads();
            for (int j = tid; j < SHdim; j += 256) sh[j] = eluf(sh[j]);
            __syncthreads();
            vecmat(sh, Wx2, bx2, sxrp, SHdim, SPdim); __syncthreads();
            // g_mu = mlp_mu(gx_retr)
            vecmat(sr, Wmu1, bmu1, sh, SPdim, 2 * Gdim); __syncthreads();
            for (int j = tid; j < 2 * Gdim; j += 256) sh[j] = eluf(sh[j]);
            __syncthreads();
            vecmat(sh, Wmu2, bmu2, sgmu, 2 * Gdim, Gdim); __syncthreads();
            // err = mean((x_retr_pred - xp)^2)
            float dd = sxrp[tid] - sxp[tid];
            sred[tid] = dd * dd; __syncthreads();
            for (int s = 128; s > 0; s >>= 1) { if (tid < s) sred[tid] += sred[tid + s]; __syncthreads(); }
            const float err = sred[0] * (1.f / SPdim);
            __syncthreads();
            // gate = sigmoid(mlp_d(concat[g_mu, g_gen, err]))
            sgi[tid] = (tid < Gdim) ? sgmu[tid] : sggen[tid - Gdim];
            if (tid == 0) sgi[2 * Gdim] = err;
            __syncthreads();
            vecmat(sgi, Wd1, bd1, sh, 2 * Gdim + 1, 10 * Gdim); __syncthreads();
            for (int j = tid; j < 10 * Gdim; j += 256) sh[j] = eluf(sh[j]);
            __syncthreads();
            vecmat(sh, Wd2, bd2, sgate, 10 * Gdim, Gdim); __syncthreads();
            if (tid < Gdim) {
                const float gt = sigm(sgate[tid]);
                sgint[tid] = sggen[tid] + (sgmu[tid] - sggen[tid]) * 0.5f * gt;  // P2G_SCALE
            }
            __syncthreads();
            vecmat(sgint, Wp, nullptr, sgproj, Gdim, SPdim); __syncthreads();
            smin[tid] = sgproj[tid] + sxp[tid]; __syncthreads();
            // x_int_pred -> out[xi]   (pre-update M)
            retrv(smin, M, sr); __syncthreads();
            vecmat(sr, Wx1, bx1, sh, SPdim, SHdim); __syncthreads();
            for (int j = tid; j < SHdim; j += 256) sh[j] = eluf(sh[j]);
            __syncthreads();
            vecmat(sh, Wx2, bx2, orow + 2 * Gdim + SPdim, SHdim, SPdim);
            if (tid < Gdim) { orow[tid] = sggen[tid]; orow[Gdim + tid] = sgint[tid]; sg[tid] = sgint[tid]; }
            __syncthreads();
        }

        // mem_update(M, S, smin)
        vecmat(smin, Wk, nullptr, sk,  SPdim, SPdim);
        vecmat(smin, Wv, nullptr, sv2, SPdim, SPdim);
        __syncthreads();
        retrv(sk, M, srk);
        __syncthreads();
        for (int i = tid; i < SPdim * SPdim; i += 256) {
            const int d = i >> 8, e = i & 255;
            const float grad = sk[d] * (srk[e] - sv2[e]);
            const float snew = 0.9f * S[i] + grad;        // ETA = 0.9
            S[i] = snew;
            M[i] = 0.9f * M[i] - 0.1f * snew;             // (1-ALPHA), THETA
        }
        __syncthreads();
    }
}

// ---------------------------------------------------------------------------
extern "C" void kernel_launch(void* const* d_in, const int* in_sizes, int n_in,
                              void* d_out, int out_size, void* d_ws, size_t ws_size,
                              hipStream_t stream)
{
    const float* act  = (const float*)d_in[0];
    const float* sen  = (const float*)d_in[1];
    const float* Wa1  = (const float*)d_in[2];
    const float* Wa2  = (const float*)d_in[3];
    const float* g0   = (const float*)d_in[4];
    const float* We1  = (const float*)d_in[5];
    const float* be1  = (const float*)d_in[6];
    const float* ln_g = (const float*)d_in[7];
    const float* ln_b = (const float*)d_in[8];
    const float* We2  = (const float*)d_in[9];
    const float* be2  = (const float*)d_in[10];
    const float* Wp   = (const float*)d_in[11];
    const float* Wmu1 = (const float*)d_in[12];
    const float* bmu1 = (const float*)d_in[13];
    const float* Wmu2 = (const float*)d_in[14];
    const float* bmu2 = (const float*)d_in[15];
    const float* Wd1  = (const float*)d_in[16];
    const float* bd1  = (const float*)d_in[17];
    const float* Wd2  = (const float*)d_in[18];
    const float* bd2  = (const float*)d_in[19];
    const float* Wx1  = (const float*)d_in[20];
    const float* bx1  = (const float*)d_in[21];
    const float* Wx2  = (const float*)d_in[22];
    const float* bx2  = (const float*)d_in[23];
    const float* Wk   = (const float*)d_in[24];
    const float* Wv   = (const float*)d_in[25];
    const float* M0   = (const float*)d_in[26];
    const float* S0   = (const float*)d_in[27];
    float* out = (float*)d_out;

    uint8_t* ws = (uint8_t*)d_ws;
    size_t off = 0;
    auto take = [&](size_t bytes) -> void* {
        off = (off + 255) & ~(size_t)255;
        void* p = ws + off; off += bytes; return p;
    };
    __bf16* sen_bf  = (__bf16*)take((size_t)BT * SVdim * 2);       //   4 MB
    __bf16* We1T    = (__bf16*)take((size_t)SPdim * SVdim * 2);
    float*  h1      = (float* )take((size_t)BT * SPdim * 4);       //   4 MB
    __bf16* ln_bf   = (__bf16*)take((size_t)BT * SPdim * 2);
    __bf16* We2T    = (__bf16*)take((size_t)SPdim * SPdim * 2);
    float*  xseq    = (float* )take((size_t)BT * SPdim * 4);       //   4 MB
    __bf16* h2_bf   = (__bf16*)take((size_t)BT * G3dim * 2);       //   3 MB
    __bf16* Wa2T    = (__bf16*)take((size_t)GGdim * G3dim * 2);    //  12 MB
    __bf16* Wt_bf   = (__bf16*)take((size_t)BT * GGdim * 2);       // 134 MB
    float*  Mw      = (float* )take((size_t)Bdim * SPdim * SPdim * 4); // 16.8 MB
    float*  Sw      = (float* )take((size_t)Bdim * SPdim * SPdim * 4); // 16.8 MB

    // ---- parallel phase: x_seq -------------------------------------------
    k_cvt_bf16<<<(BT * SVdim + 255) / 256, 256, 0, stream>>>(sen, sen_bf, BT * SVdim);
    k_transpose_bf16<<<(SVdim * SPdim + 255) / 256, 256, 0, stream>>>(We1, We1T, SVdim, SPdim);
    k_gemm_bf16<<<dim3(BT / 64, SPdim / 64), 128, 0, stream>>>(
        sen_bf, We1T, be1, h1, nullptr, BT, SPdim, SVdim);
    k_ln_elu<<<BT, 256, 0, stream>>>(h1, ln_g, ln_b, ln_bf);
    k_transpose_bf16<<<(SPdim * SPdim + 255) / 256, 256, 0, stream>>>(We2, We2T, SPdim, SPdim);
    k_gemm_bf16<<<dim3(BT / 64, SPdim / 64), 128, 0, stream>>>(
        ln_bf, We2T, be2, xseq, nullptr, BT, SPdim, SPdim);

    // ---- parallel phase: hoisted Wt = tanh(act@Wa1)@Wa2 (51.5 GFLOP) -----
    k_tanh_a<<<dim3(BT, (G3dim + 127) / 128), 128, 0, stream>>>(act, Wa1, h2_bf);
    k_transpose_bf16<<<(G3dim * GGdim + 255) / 256, 256, 0, stream>>>(Wa2, Wa2T, G3dim, GGdim);
    k_gemm_bf16<<<dim3(BT / 64, GGdim / 64), 128, 0, stream>>>(
        h2_bf, Wa2T, nullptr, nullptr, Wt_bf, BT, GGdim, G3dim);

    // ---- sequential phase: per-batch scan, M/S L2-resident ---------------
    k_scan<<<Bdim, 256, 0, stream>>>(
        Wt_bf, xseq, g0, Wp, Wmu1, bmu1, Wmu2, bmu2,
        Wd1, bd1, Wd2, bd2, Wx1, bx1, Wx2, bx2, Wk, Wv,
        M0, S0, Mw, Sw, out);

    (void)in_sizes; (void)n_in; (void)out_size; (void)ws_size;
}
